// DotProductAttention_41300405518967
// MI455X (gfx1250) — matrix operand order repaired and checked by reading, not compiled, not run on hardware
//
#include <hip/hip_runtime.h>

typedef __attribute__((ext_vector_type(16))) _Float16 v16h;
typedef __attribute__((ext_vector_type(8)))  float    v8f;

#define SQ 2048
#define BB 2
#define HH 16
#define DD 64
#define RS (BB*HH*DD)   // 2048 floats between consecutive sq rows

// Flash attention forward, f16 WMMA with f32 accumulation.
// One wave handles 16 query rows of one (b,h); streams keys in tiles of 32.
__global__ __launch_bounds__(256) void fa_fwd_kernel(
    const float* __restrict__ Q, const float* __restrict__ Km,
    const float* __restrict__ Vm, const unsigned char* __restrict__ mask,
    float* __restrict__ out)
{
  const int lane = threadIdx.x & 31;
  const int wave = threadIdx.x >> 5;
  const int l16  = lane & 15;
  const int half = lane >> 4;            // 0 or 1 (lane half)
  const int h = blockIdx.y;
  const int b = blockIdx.z;
  const int q0 = blockIdx.x * 128 + wave * 16;
  const int headoff = (b * HH + h) * DD;

  // ---- Q^T as B-operand of GEMM1 (held in registers for the whole kernel) ----
  // B layout (32x16, 16-bit): lane holds column q=l16; half0 -> K(d)=0..15,
  // half1 -> K(d)=16..31 of each 32-wide d-chunk, in order.
  const int qrow = q0 + l16;
  const float* qptr = Q + (size_t)qrow * RS + headoff;
  v16h qB[2];
#pragma unroll
  for (int c = 0; c < 2; ++c) {
    const float* p = qptr + c*32 + half*16;
#pragma unroll
    for (int e = 0; e < 16; ++e) qB[c][e] = (_Float16)p[e];
  }

  const unsigned char* mrow = mask + ((size_t)(b*HH + h) * SQ + qrow) * SQ;

  float mval = -3.0e38f;   // running row max (per q = l16, replicated in lane pair)
  float lval = 0.0f;       // running row sum
  v8f o[4];                // O acc: o[c][r] = O[q=r+8*half][dv=c*16+l16]
#pragma unroll
  for (int c = 0; c < 4; ++c) o[c] = (v8f){0,0,0,0,0,0,0,0};

  for (int kb = 0; kb < SQ; kb += 32) {
    // ---- GEMM1: ST = K_tile * Q^T, two 16-key subtiles --------------------
    // A layout (16x32, 16-bit): row = key = kb+sub*16+l16;
    // half0: d-elems 0..15 +32c, half1: 16..31 +32c (contiguous, matches qB).
    v8f st[2];
#pragma unroll
    for (int sub = 0; sub < 2; ++sub) {
      const float* kptr = Km + (size_t)(kb + sub*16 + l16) * RS + headoff;
      v8f acc = (v8f){0,0,0,0,0,0,0,0};
#pragma unroll
      for (int c = 0; c < 2; ++c) {
        v16h a;
        const float* g0 = kptr + c*32 + half*16;
        const float* g1 = kptr + c*32 + half*16 + 8;
#pragma unroll
        for (int e = 0; e < 8; ++e) { a[e] = (_Float16)g0[e]; a[e+8] = (_Float16)g1[e]; }
        acc = __builtin_amdgcn_wmma_f32_16x16x32_f16(
                  false, a, false, qB[c], (short)0, acc, false, false);
      }
      st[sub] = acc;
    }

    // ---- scale by 1/sqrt(D)=1/8, apply mask -------------------------------
    // Lane's 8 keys per subtile are contiguous: one 8-byte mask load each.
    unsigned long long mbA = *(const unsigned long long*)(mrow + kb + 8*half);
    unsigned long long mbB = *(const unsigned long long*)(mrow + kb + 16 + 8*half);
    float s0[8], s1[8];
#pragma unroll
    for (int r = 0; r < 8; ++r) {
      s0[r] = ((mbA >> (8*r)) & 0xffull) ? -10000.0f : st[0][r] * 0.125f;
      s1[r] = ((mbB >> (8*r)) & 0xffull) ? -10000.0f : st[1][r] * 0.125f;
    }

    // ---- online softmax (per q = l16; partner lane is L^16) ---------------
    float mt = -3.0e38f;
#pragma unroll
    for (int r = 0; r < 8; ++r) { mt = fmaxf(mt, s0[r]); mt = fmaxf(mt, s1[r]); }
    mt = fmaxf(mt, __shfl_xor(mt, 16, 32));
    const float mnew  = fmaxf(mval, mt);
    const float scale = __expf(mval - mnew);
    mval = mnew;

    float ls = 0.0f;
    v16h aP;  // P directly in A-operand layout for GEMM2 (the S^T trick)
#pragma unroll
    for (int r = 0; r < 8; ++r) {
      const float p0 = __expf(s0[r] - mnew);
      const float p1 = __expf(s1[r] - mnew);
      ls += p0 + p1;
      aP[r]   = (_Float16)p0;   // keys kb + 8*half + r
      aP[r+8] = (_Float16)p1;   // keys kb + 16 + 8*half + r
    }
    ls += __shfl_xor(ls, 16, 32);
    lval = lval * scale + ls;

    // ---- V tile as B-operand: lane holds column dv=c*16+l16;
    //      half0 -> keys kb+{0..7,16..23}, half1 -> keys kb+{8..15,24..31} ---
    v16h vB[4];
#pragma unroll
    for (int c = 0; c < 4; ++c) {
      const float* vcol = Vm + (size_t)(kb + half*8) * RS + headoff + c*16 + l16;
#pragma unroll
      for (int e = 0; e < 16; ++e) vB[c][e] = (_Float16)vcol[(size_t)(e + (e >= 8 ? 8 : 0)) * RS];
    }

    // ---- rescale O (rows are q=r+8*half -> fetch scale from lane r+8*half) -
    float bsc[8];
#pragma unroll
    for (int r = 0; r < 8; ++r) bsc[r] = __shfl(scale, r + 8*half, 32);
#pragma unroll
    for (int c = 0; c < 4; ++c) {
#pragma unroll
      for (int r = 0; r < 8; ++r) o[c][r] *= bsc[r];
      o[c] = __builtin_amdgcn_wmma_f32_16x16x32_f16(
                 false, aP, false, vB[c], (short)0, o[c], false, false);
    }
  }

  // ---- normalize by row sum and store (coalesced 64B per row-chunk) --------
  const float inv = 1.0f / lval;
  float binv[8];
#pragma unroll
  for (int r = 0; r < 8; ++r) binv[r] = __shfl(inv, r + 8*half, 32);
#pragma unroll
  for (int c = 0; c < 4; ++c) {
#pragma unroll
    for (int r = 0; r < 8; ++r) {
      out[(size_t)(q0 + r + 8*half) * RS + headoff + c*16 + l16] = o[c][r] * binv[r];
    }
  }
}

extern "C" void kernel_launch(void* const* d_in, const int* in_sizes, int n_in,
                              void* d_out, int out_size, void* d_ws, size_t ws_size,
                              hipStream_t stream) {
  const float* Q = (const float*)d_in[0];
  const float* K = (const float*)d_in[1];
  const float* V = (const float*)d_in[2];
  const unsigned char* mask = (const unsigned char*)d_in[3];  // jnp bool -> 1 byte
  float* out = (float*)d_out;
  (void)in_sizes; (void)n_in; (void)out_size; (void)d_ws; (void)ws_size;

  dim3 grid(SQ / 128, HH, BB);   // 16 x 16 x 2 = 512 workgroups
  fa_fwd_kernel<<<grid, 256, 0, stream>>>(Q, K, V, mask, out);
}
